// SpatialAwareDecoder_67027259621778
// MI455X (gfx1250) — matrix operand order repaired and checked by reading, not compiled, not run on hardware
//
#include <hip/hip_runtime.h>
#include <math.h>

// ---------------------------------------------------------------------------
// SpatialAwareDecoder for MI455X (gfx1250, wave32, WMMA)
// Gather + Fourier-encode -> [1087 -> 256 -> 256 -> 14] MLP with LN/ReLU,
// all GEMMs on v_wmma_f32_16x16x32_f16 (f16 in, f32 accumulate).
// 32 rows per block: ~32 FLOP per byte of L2 weight traffic, A-fragment
// reuse x4 per ds load. LN f32 staging overlays the (dead) input LDS tile.
// ---------------------------------------------------------------------------

typedef __attribute__((ext_vector_type(16))) _Float16 v16h;
typedef __attribute__((ext_vector_type(8)))  _Float16 v8h;
typedef __attribute__((ext_vector_type(8)))  float    v8f;

#define TOKEN_DIM   1024
#define HIDDEN      256
#define NUM_FREQS   10
#define IN_DIM      1087          // 1024 + 3*(2*10+1)
#define K1PAD       1088          // = 34 * 32
#define BATCH       2
#define TSEQ        512
#define NPTS        40000
#define ROWS_TOTAL  (BATCH * NPTS)
#define BLOCK_ROWS  32            // 2500 blocks, 2 m-tiles of 16

#define NT1 16                    // 256 / 16 output col tiles, layer 1
#define KB1 34                    // 1088 / 32 K-blocks
#define NT2 16
#define KB2 8                     // 256 / 32
#define KB3 8

#define FRAG_BASE_W1 0
#define FRAG_BASE_W2 (NT1 * KB1)                  // 544
#define FRAG_BASE_W3 (FRAG_BASE_W2 + NT2 * KB2)   // 672
#define TOTAL_FRAGS  (FRAG_BASE_W3 + KB3)         // 680
// ws usage: TOTAL_FRAGS * 512 halves = 696,320 bytes

#define A_STRIDE (K1PAD + 8)      // 1096 halves: 16B aligned, bank-friendly
#define H_STRIDE (HIDDEN + 8)     // 264 halves
#define S_STRIDE (HIDDEN + 1)     // 257 floats (odd dword stride)
#define SMEM_A_BYTES (BLOCK_ROWS * A_STRIDE * 2)          // 70,144
#define SMEM_S_BYTES (BLOCK_ROWS * S_STRIDE * 4)          // 32,896 (overlaid)

// ---------------------------------------------------------------------------
// Weight prep: swizzle W1/W2/W3 (f32, row-major [K][N]) into per-lane
// B-matrix fragments for v_wmma_f32_16x16x32_f16:
//   element i of lane L in k-block kb, n-tile t holds
//   B[K = kb*32 + 8*(L/16) + (i%8) + 16*(i/8)][N = t*16 + (L%16)]   (zero-padded OOB).
// One contiguous 32B v16h per lane per fragment at GEMM time.
// ---------------------------------------------------------------------------
__global__ void prep_weights(const float* __restrict__ W1,
                             const float* __restrict__ W2,
                             const float* __restrict__ W3,
                             _Float16* __restrict__ wsf) {
  const int fb  = blockIdx.x;        // fragment-block id
  const int tid = threadIdx.x;       // 0..511 -> lane*16 + i
  const int lane = tid >> 4;         // 0..31
  const int i    = tid & 15;         // 0..15

  const float* src;
  int Kdim, Ndim, t, kb;
  if (fb < FRAG_BASE_W2)      { src = W1; Kdim = IN_DIM; Ndim = HIDDEN;
                                t = fb / KB1; kb = fb % KB1; }
  else if (fb < FRAG_BASE_W3) { int f = fb - FRAG_BASE_W2;
                                src = W2; Kdim = HIDDEN; Ndim = HIDDEN;
                                t = f / KB2; kb = f % KB2; }
  else                        { int f = fb - FRAG_BASE_W3;
                                src = W3; Kdim = HIDDEN; Ndim = 14;
                                t = 0; kb = f; }

  const int K = kb * 32 + (lane >> 4) * 8 + (i & 7) + 16 * (i >> 3);
  const int n = t * 16 + (lane & 15);
  const float v = (K < Kdim && n < Ndim) ? src[(size_t)K * Ndim + n] : 0.0f;
  wsf[(size_t)fb * 512 + tid] = (_Float16)v;
}

// ---------------------------------------------------------------------------
// Row-wise LayerNorm + ReLU over 256 cols (f32 staging -> f16 A-operand LDS).
// 4 rows per wave, wave32 shfl_xor reduction.
// ---------------------------------------------------------------------------
__device__ __forceinline__ void ln_relu(int wave, int lane,
                                        float (*Sf)[S_STRIDE],
                                        _Float16 (*Hf)[H_STRIDE],
                                        const float* __restrict__ g,
                                        const float* __restrict__ be) {
  for (int rr = 0; rr < 4; ++rr) {
    const int r = wave * 4 + rr;
    float s = 0.0f, ss = 0.0f;
    float vals[8];
#pragma unroll
    for (int k = 0; k < 8; ++k) {
      const float x = Sf[r][lane + 32 * k];
      vals[k] = x; s += x; ss += x * x;
    }
#pragma unroll
    for (int m = 16; m >= 1; m >>= 1) {
      s  += __shfl_xor(s,  m, 32);
      ss += __shfl_xor(ss, m, 32);
    }
    const float mu   = s * (1.0f / HIDDEN);
    const float var  = ss * (1.0f / HIDDEN) - mu * mu;
    const float rstd = rsqrtf(var + 1e-5f);
#pragma unroll
    for (int k = 0; k < 8; ++k) {
      const int c = lane + 32 * k;
      float y = (vals[k] - mu) * rstd * g[c] + be[c];
      y = fmaxf(y, 0.0f);
      Hf[r][c] = (_Float16)y;
    }
  }
}

// ---------------------------------------------------------------------------
// Main fused kernel: 256 threads (8 waves), 32 rows per block.
// Wave w: m-tile mt = w>>2 (16 rows), n-tiles (w&3)*4 .. +3 (64 cols).
// ---------------------------------------------------------------------------
__global__ void __launch_bounds__(256) decoder_kernel(
    const float* __restrict__ tokens, const float* __restrict__ anchors,
    const int* __restrict__ ids,
    const float* __restrict__ b1, const float* __restrict__ g1, const float* __restrict__ be1,
    const float* __restrict__ b2, const float* __restrict__ g2, const float* __restrict__ be2,
    const float* __restrict__ b3,
    const _Float16* __restrict__ wsf,
    float* __restrict__ out) {
  // Af (f16 input tile) and Sf (f32 LN staging) share storage: Af is dead
  // once GEMM accumulation has read it (barrier-separated below).
  __shared__ __align__(16) unsigned char smemAS[SMEM_A_BYTES]; // >= SMEM_S_BYTES
  __shared__ __align__(16) _Float16 Hf[BLOCK_ROWS][H_STRIDE];  // 16,896 B
  __shared__ float Raw[BLOCK_ROWS][17];                        //  2,176 B
  _Float16 (*Af)[A_STRIDE] = reinterpret_cast<_Float16 (*)[A_STRIDE]>(smemAS);
  float    (*Sf)[S_STRIDE] = reinterpret_cast<float    (*)[S_STRIDE]>(smemAS);

  const int tid  = threadIdx.x;
  const int lane = tid & 31;
  const int wave = tid >> 5;
  const int half = lane >> 4;        // 0/1: which 16-lane half of the wave
  const int l16  = lane & 15;
  const int mt   = wave >> 2;        // m-tile (0/1)
  const int nq   = wave & 3;         // n-quad: tiles nq*4 .. nq*4+3
  const int arow = mt * 16 + l16;    // A-fragment source row
  const int rowBase = blockIdx.x * BLOCK_ROWS;
  const v16h* __restrict__ wf = (const v16h*)wsf;

  // ---- Phase 0: gather tokens + Fourier encode anchors -> Af (f16) --------
  for (int r = 0; r < BLOCK_ROWS; ++r) {
    const int p  = rowBase + r;
    const int bb = p / NPTS;
    const int nn = p - bb * NPTS;
    const int idx = ids[(size_t)bb * NPTS + nn];
    const float* __restrict__ tok = tokens  + ((size_t)bb * TSEQ + idx) * TOKEN_DIM;
    const float* __restrict__ anc = anchors + ((size_t)bb * TSEQ + idx) * 3;
    for (int c = tid; c < K1PAD; c += 256) {
      float v;
      if (c < TOKEN_DIM) {
        v = tok[c];
      } else if (c < IN_DIM) {
        const int f = c - TOKEN_DIM;
        const int d = f / 21, j = f - d * 21;         // [x, sin*10, cos*10] per dim
        const float xd = anc[d];
        if (j == 0)              v = xd;
        else if (j <= NUM_FREQS) v = sinf(xd * (float)(1 << (j - 1)));
        else                     v = cosf(xd * (float)(1 << (j - 1 - NUM_FREQS)));
      } else {
        v = 0.0f;                                      // K padding col 1087
      }
      Af[r][c] = (_Float16)v;
    }
  }
  __syncthreads();

  // ---- Phase 1: GEMM1 (32x1088 @ 1088x256), 4 n-tiles per wave ------------
  {
    v8f acc[4] = {{}, {}, {}, {}};
    for (int kb = 0; kb < KB1; ++kb) {
      union { v16h v; v8h h[2]; } a;
      const _Float16* ap = &Af[arow][kb * 32 + 8 * half];
      a.h[0] = *(const v8h*)(ap);        // K = kb*32 + 8*half + [0..7]
      a.h[1] = *(const v8h*)(ap + 16);   // K = kb*32 + 8*half + 16 + [0..7]
#pragma unroll
      for (int t = 0; t < 4; ++t) {
        const int tile = nq * 4 + t;
        const v16h bf = wf[(size_t)(FRAG_BASE_W1 + tile * KB1 + kb) * 32 + lane];
        acc[t] = __builtin_amdgcn_wmma_f32_16x16x32_f16(false, a.v, false, bf,
                                                        (short)0, acc[t], false, false);
      }
    }
    __syncthreads();                     // last Af read before Sf overlay write
#pragma unroll
    for (int t = 0; t < 4; ++t) {
#pragma unroll
      for (int v = 0; v < 8; ++v) {      // C layout: M = v + 8*half, N = l16
        const int m = mt * 16 + 8 * half + v;
        const int c = (nq * 4 + t) * 16 + l16;
        Sf[m][c] = acc[t][v] + b1[c];
      }
    }
  }
  __syncthreads();

  // ---- Phase 2: LN1 + ReLU -> Hf (f16) ------------------------------------
  ln_relu(wave, lane, Sf, Hf, g1, be1);
  __syncthreads();

  // ---- Phase 3: GEMM2 (32x256 @ 256x256) ----------------------------------
  {
    v8f acc[4] = {{}, {}, {}, {}};
    for (int kb = 0; kb < KB2; ++kb) {
      union { v16h v; v8h h[2]; } a;
      const _Float16* ap = &Hf[arow][kb * 32 + 8 * half];
      a.h[0] = *(const v8h*)(ap);
      a.h[1] = *(const v8h*)(ap + 16);
#pragma unroll
      for (int t = 0; t < 4; ++t) {
        const int tile = nq * 4 + t;
        const v16h bf = wf[(size_t)(FRAG_BASE_W2 + tile * KB2 + kb) * 32 + lane];
        acc[t] = __builtin_amdgcn_wmma_f32_16x16x32_f16(false, a.v, false, bf,
                                                        (short)0, acc[t], false, false);
      }
    }
    // Sf (overlay) is idle since LN1's barrier; Hf readers untouched.
#pragma unroll
    for (int t = 0; t < 4; ++t) {
#pragma unroll
      for (int v = 0; v < 8; ++v) {
        const int m = mt * 16 + 8 * half + v;
        const int c = (nq * 4 + t) * 16 + l16;
        Sf[m][c] = acc[t][v] + b2[c];
      }
    }
  }
  __syncthreads();

  // ---- Phase 4: LN2 + ReLU -> Hf ------------------------------------------
  ln_relu(wave, lane, Sf, Hf, g2, be2);
  __syncthreads();

  // ---- Phase 5: GEMM3 (32x256 @ 256x14, padded N=16), waves 0..1 ----------
  if (wave < 2) {
    v8f acc = {};
    for (int kb = 0; kb < KB3; ++kb) {
      union { v16h v; v8h h[2]; } a;
      const _Float16* ap = &Hf[wave * 16 + l16][kb * 32 + 8 * half];
      a.h[0] = *(const v8h*)(ap);
      a.h[1] = *(const v8h*)(ap + 16);
      const v16h bf = wf[(size_t)(FRAG_BASE_W3 + kb) * 32 + lane];
      acc = __builtin_amdgcn_wmma_f32_16x16x32_f16(false, a.v, false, bf,
                                                   (short)0, acc, false, false);
    }
#pragma unroll
    for (int v = 0; v < 8; ++v) {
      const int m = wave * 16 + 8 * half + v;
      const int n = l16;
      Raw[m][n] = acc[v] + (n < 14 ? b3[n] : 0.0f);
    }
  }
  __syncthreads();

  // ---- Phase 6: heads (clip / sigmoid / exp / quat-normalize) + store -----
  for (int e = tid; e < BLOCK_ROWS * 14; e += 256) {
    const int r = e / 14, j = e - r * 14;
    const float x = Raw[r][j];
    float y;
    if (j < 3)       y = x;                                   // pos
    else if (j < 6)  y = fminf(fmaxf(x, 0.0f), 1.0f);         // color
    else if (j == 6) y = 1.0f / (1.0f + expf(-x));            // opacity
    else if (j < 10) y = expf(x);                             // scale
    else {                                                    // quaternion
      const float q0 = Raw[r][10], q1 = Raw[r][11];
      const float q2 = Raw[r][12], q3 = Raw[r][13];
      const float nrm = sqrtf(q0 * q0 + q1 * q1 + q2 * q2 + q3 * q3);
      y = x / fmaxf(nrm, 1e-12f);
    }
    const int p  = rowBase + r;
    const int bb = p / NPTS;
    const int nn = p - bb * NPTS;
    out[(size_t)bb * ((size_t)NPTS * 14) + (size_t)nn * 14 + j] = y;
  }
}

// ---------------------------------------------------------------------------
extern "C" void kernel_launch(void* const* d_in, const int* in_sizes, int n_in,
                              void* d_out, int out_size, void* d_ws, size_t ws_size,
                              hipStream_t stream) {
  (void)in_sizes; (void)n_in; (void)out_size;
  const float* tokens  = (const float*)d_in[0];
  const float* anchors = (const float*)d_in[1];
  const int*   ids     = (const int*)d_in[2];
  const float* W1  = (const float*)d_in[3];
  const float* b1  = (const float*)d_in[4];
  const float* g1  = (const float*)d_in[5];
  const float* be1 = (const float*)d_in[6];
  const float* W2  = (const float*)d_in[7];
  const float* b2  = (const float*)d_in[8];
  const float* g2  = (const float*)d_in[9];
  const float* be2 = (const float*)d_in[10];
  const float* W3  = (const float*)d_in[11];
  const float* b3  = (const float*)d_in[12];
  float* out = (float*)d_out;
  _Float16* wsf = (_Float16*)d_ws;

  if (ws_size < (size_t)TOTAL_FRAGS * 512 * sizeof(_Float16)) return;

  prep_weights<<<TOTAL_FRAGS, 512, 0, stream>>>(W1, W2, W3, wsf);
  decoder_kernel<<<ROWS_TOTAL / BLOCK_ROWS, 256, 0, stream>>>(
      tokens, anchors, ids, b1, g1, be1, b2, g2, be2, b3, wsf, out);
}
